// GraphUNetLayer_43336220016827
// MI455X (gfx1250) — matrix operand order repaired and checked by reading, not compiled, not run on hardware
//
#include <hip/hip_runtime.h>
#include <hip/hip_bf16.h>
#include <math.h>

#define BB 16
#define NN 1024
#define CIN 256
#define COUT 256

typedef __attribute__((ext_vector_type(8)))  __bf16 bf16x8;
typedef __attribute__((ext_vector_type(16))) __bf16 bf16x16;
typedef __attribute__((ext_vector_type(8)))  float  floatx8;
typedef __attribute__((ext_vector_type(4)))  unsigned int u32x4;
typedef __attribute__((ext_vector_type(4)))  int i32x4;
typedef __attribute__((ext_vector_type(8)))  int i32x8;

static __device__ __forceinline__ __bf16 f32_to_bf16_rne(float f) {
    unsigned u = __builtin_bit_cast(unsigned, f);
    unsigned r = u + 0x7FFFu + ((u >> 16) & 1u);
    unsigned short s = (unsigned short)(r >> 16);
    return __builtin_bit_cast(__bf16, s);
}

// ---------------------------------------------------------------- converters
__global__ void k_f32_to_bf16(const float* __restrict__ src,
                              __bf16* __restrict__ dst, int n) {
    int i = blockIdx.x * 256 + threadIdx.x;
    if (i < n) dst[i] = f32_to_bf16_rne(src[i]);
}

// W[C_IN][C_OUT] -> Wt[C_OUT][C_IN] (K-major for WMMA B-fragments)
__global__ void k_transpose_w(const float* __restrict__ W,
                              __bf16* __restrict__ Wt) {
    int i = blockIdx.x * 256 + threadIdx.x;   // i < 65536
    int r = i >> 8;        // c_in  (K)
    int c = i & 255;       // c_out (N)
    Wt[(size_t)c * CIN + r] = f32_to_bf16_rne(W[i]);
}

// ------------------------------------------------------------------- GEMM 1
// h0t[b][c][n] = bf16( sum_k x[b][n][k] * W[k][c] )   (stored K-major for GEMM2)
__global__ void k_gemm1(const __bf16* __restrict__ xb,
                        const __bf16* __restrict__ wt,
                        __bf16* __restrict__ h0t) {
    const int lane = threadIdx.x & 31;
    const int wave = threadIdx.x >> 5;
    const int tile = blockIdx.x * 8 + wave;     // 16384 tiles total
    const int bat  = tile >> 10;                // 1024 tiles per batch
    const int rem  = tile & 1023;
    const int tm   = rem >> 4;                  // 0..63  node-row tile
    const int tn   = rem & 15;                  // 0..15  c_out tile
    const int half = lane >> 4;
    const int l16  = lane & 15;
    const int row  = tm * 16 + l16;             // A row (M)
    const int col  = tn * 16 + l16;             // B col (N)

    const __bf16* __restrict__ Arow = xb + ((size_t)bat * NN + row) * CIN;
    const __bf16* __restrict__ Brow = wt + (size_t)col * CIN;

    floatx8 acc = {};
    #pragma unroll
    for (int kk = 0; kk < CIN; kk += 32) {
        bf16x8  a0 = *(const bf16x8*) (Arow + kk + half * 16);
        bf16x8  a1 = *(const bf16x8*) (Arow + kk + half * 16 + 8);
        bf16x16 af = __builtin_shufflevector(a0, a1,
                       0,1,2,3,4,5,6,7,8,9,10,11,12,13,14,15);
        bf16x16 bf = *(const bf16x16*)(Brow + kk + half * 16);
        acc = __builtin_amdgcn_wmma_f32_16x16x32_bf16(
                  false, af, false, bf, (short)0, acc, false, false);
    }
    bf16x8 o;
    #pragma unroll
    for (int r = 0; r < 8; ++r) o[r] = f32_to_bf16_rne(acc[r]);
    *(bf16x8*)(h0t + ((size_t)bat * COUT + col) * NN + tm * 16 + half * 8) = o;
}

// --------------------------------------------------------- TDM availability
#if defined(__HIP_DEVICE_COMPILE__) && \
    __has_builtin(__builtin_amdgcn_tensor_load_to_lds) && \
    __has_builtin(__builtin_amdgcn_s_wait_tensorcnt)
#define HAVE_TDM 1
#else
#define HAVE_TDM 0
#endif

// LDS tile layout: 64 rows x 1024 bf16, padded: 16B inserted after every 1024B
// -> row stride 2080B (520 dwords) keeps 16B-chunk alignment and breaks the
//    2048B power-of-two bank pattern.
#define LDS_ROW_STRIDE 2080
#define LDS_TILE_BYTES (64 * LDS_ROW_STRIDE)   // 133120 ; x2 = 266240 <= 320KB

static __device__ __forceinline__ const __bf16* lds_chunk(const char* base,
                                                          int row, int k) {
    // element (row, k): byte = row*2080 + 2*k + 16*(k/512); chunks of 8/16
    // elements at k%8==0 never straddle a 512-element pad boundary.
    return (const __bf16*)(base + row * LDS_ROW_STRIDE + 2 * k + ((k >> 9) << 4));
}

#if HAVE_TDM
// One-shot DMA of a 64x1024 bf16 band (2D tile) into LDS with padding.
static __device__ __forceinline__ void tdm_load_band(const void* gsrc,
                                                     unsigned lds_byte_off) {
    unsigned long long ga = (unsigned long long)gsrc;
    // D# group0: count=1 | lds_addr | global_addr[56:0] | type=2
    u32x4 g0 = { 1u,
                 lds_byte_off,
                 (unsigned)(ga & 0xffffffffu),
                 (unsigned)((ga >> 32) & 0x01ffffffu) | (2u << 30) };
    // D# group1:
    //  w0: data_size=1(2B)<<16 | pad_enable<<20 | pad_interval=7(256dw)<<22
    //      | pad_amount=3(4dw=16B)<<25
    //  tensor_dim0=1024 [79:48], tensor_dim1=64 [111:80]
    //  tile_dim0=1024 [127:112], tile_dim1=64 [143:128], tile_dim2=0
    //  tensor_dim0_stride=1024 [207:160], tensor_dim1_stride=0
    i32x8 g1 = { (int)((1u << 16) | (1u << 20) | (7u << 22) | (3u << 25)),
                 (int)(1024u << 16),   // tensor_dim0[15:0] in [31:16]
                 (int)(64u << 16),     // tensor_dim0[31:16]=0 | tensor_dim1[15:0]
                 (int)(1024u << 16),   // tensor_dim1[31:16]=0 | tile_dim0
                 (int)(64u),           // tile_dim1 | tile_dim2=0
                 (int)(1024u),         // tensor_dim0_stride[31:0]
                 0, 0 };
    i32x4 z4 = { 0, 0, 0, 0 };
#if __clang_major__ >= 23
    i32x8 z8 = { 0, 0, 0, 0, 0, 0, 0, 0 };
    __builtin_amdgcn_tensor_load_to_lds(g0, g1, z4, z4, z8, 0);
#else
    __builtin_amdgcn_tensor_load_to_lds(g0, g1, z4, z4, 0);
#endif
}
#endif

// ------------------------------------------------------------------- GEMM 2
// h[b][n][c] = sum_m adj[b][n][m] * h0[b][m][c] + bias[c]
// Block = 64x64 output tile; A/B K-bands staged whole in LDS (TDM DMA).
__global__ void k_gemm2(const __bf16* __restrict__ adjb,
                        const __bf16* __restrict__ h0t,
                        const float* __restrict__ bias,
                        float* __restrict__ h) {
    extern __shared__ char smem[];
    char* ldsA = smem;                       // 64 rows (nodes)  x 1024 K
    char* ldsB = smem + LDS_TILE_BYTES;      // 64 rows (c_out)  x 1024 K

    const int tid  = threadIdx.x;
    const int lane = tid & 31;
    const int wave = tid >> 5;
    const int bat  = blockIdx.x >> 6;        // 64 blocks per batch
    const int rem  = blockIdx.x & 63;
    const int rb   = rem >> 2;               // 0..15  row block (64 nodes)
    const int cb   = rem & 3;                // 0..3   col block (64 c_out)
    const int row0 = rb * 64;
    const int col0 = cb * 64;

    const __bf16* gA = adjb + ((size_t)bat * NN + row0) * NN;
    const __bf16* gB = h0t  + ((size_t)bat * COUT + col0) * NN;

#if HAVE_TDM
    if (wave == 0) {
        tdm_load_band(gA, (unsigned)(size_t)ldsA);
        tdm_load_band(gB, (unsigned)(size_t)ldsB);
        __builtin_amdgcn_s_wait_tensorcnt(0);
    }
    __syncthreads();
#else
    // cooperative copy into the same padded LDS layout
    for (int i = tid; i < 8192; i += 256) {          // 16B chunks per band
        int row = i >> 7;                            // 128 chunks per row
        int c   = i & 127;
        unsigned off = row * LDS_ROW_STRIDE + c * 16 + ((c >> 6) << 4);
        *(uint4*)(ldsA + off) = ((const uint4*)gA)[i];
        *(uint4*)(ldsB + off) = ((const uint4*)gB)[i];
    }
    __syncthreads();
#endif

    // 8 waves x 2 tiles = 16 subtiles of the 64x64 block; A-frag reused x2
    const int rt    = wave & 3;              // row subtile 0..3
    const int ct0   = (wave >> 2) << 1;      // col subtiles ct0, ct0+1
    const int half  = lane >> 4;
    const int l16   = lane & 15;
    const int arow  = rt * 16 + l16;
    const int bcol  = ct0 * 16 + l16;

    floatx8 acc0 = {}, acc1 = {};
    for (int kk = 0; kk < NN; kk += 32) {
        bf16x8  a0 = *(const bf16x8*)lds_chunk(ldsA, arow, kk + half * 16);
        bf16x8  a1 = *(const bf16x8*)lds_chunk(ldsA, arow, kk + half * 16 + 8);
        bf16x16 af = __builtin_shufflevector(a0, a1,
                       0,1,2,3,4,5,6,7,8,9,10,11,12,13,14,15);
        bf16x8  b0l = *(const bf16x8*)lds_chunk(ldsB, bcol,      kk + half * 16);
        bf16x8  b0h = *(const bf16x8*)lds_chunk(ldsB, bcol,      kk + half * 16 + 8);
        bf16x8  b1l = *(const bf16x8*)lds_chunk(ldsB, bcol + 16, kk + half * 16);
        bf16x8  b1h = *(const bf16x8*)lds_chunk(ldsB, bcol + 16, kk + half * 16 + 8);
        bf16x16 bf0 = __builtin_shufflevector(b0l, b0h,
                       0,1,2,3,4,5,6,7,8,9,10,11,12,13,14,15);
        bf16x16 bf1 = __builtin_shufflevector(b1l, b1h,
                       0,1,2,3,4,5,6,7,8,9,10,11,12,13,14,15);
        acc0 = __builtin_amdgcn_wmma_f32_16x16x32_bf16(
                   false, af, false, bf0, (short)0, acc0, false, false);
        acc1 = __builtin_amdgcn_wmma_f32_16x16x32_bf16(
                   false, af, false, bf1, (short)0, acc1, false, false);
    }
    const int gc0 = col0 + bcol;
    const float bv0 = bias[gc0];
    const float bv1 = bias[gc0 + 16];
    #pragma unroll
    for (int r = 0; r < 8; ++r) {
        size_t rowg = (size_t)bat * NN + row0 + rt * 16 + half * 8 + r;
        h[rowg * COUT + gc0]      = acc0[r] + bv0;
        h[rowg * COUT + gc0 + 16] = acc1[r] + bv1;
    }
}

// ------------------------------------------------------- projection score y
__global__ void k_score(const float* __restrict__ h,
                        const float* __restrict__ pooling,
                        float* __restrict__ y) {
    __shared__ float s1[256];
    __shared__ float s2[256];
    const int t = threadIdx.x;
    const size_t blk = blockIdx.x;              // b*N + n
    float p  = pooling[t];
    float hv = h[blk * COUT + t];
    s1[t] = hv * p;
    s2[t] = p * p;
    __syncthreads();
    for (int off = 128; off > 0; off >>= 1) {
        if (t < off) { s1[t] += s1[t + off]; s2[t] += s2[t + off]; }
        __syncthreads();
    }
    if (t == 0) y[blk] = s1[0] / sqrtf(s2[0]);
}

// ------------------------------------- per-batch argsort + top-k mask prune
__global__ void k_pool(const float* __restrict__ y,
                       const int* __restrict__ mask,
                       const int* __restrict__ n_nodes,
                       int* __restrict__ new_mask_i,
                       float* __restrict__ out_mask_f,
                       float* __restrict__ out_nn_f) {
    __shared__ float skey[NN];
    __shared__ int   sidx[NN];
    __shared__ int   sscan[NN];
    const int t = threadIdx.x;
    const int b = blockIdx.x;

    skey[t] = y[b * NN + t];
    sidx[t] = t;
    __syncthreads();

    // bitonic sort ascending (key, index)
    for (unsigned k = 2; k <= NN; k <<= 1) {
        for (unsigned j = k >> 1; j > 0; j >>= 1) {
            unsigned ixj = t ^ j;
            if (ixj > (unsigned)t) {
                bool up = ((t & k) == 0);
                float a = skey[t], c = skey[ixj];
                if (up ? (a > c) : (a < c)) {
                    skey[t] = c; skey[ixj] = a;
                    int ti = sidx[t]; sidx[t] = sidx[ixj]; sidx[ixj] = ti;
                }
            }
            __syncthreads();
        }
    }

    // gather mask in sorted order; inclusive scan; exclusive rank
    const int m = mask[b * NN + sidx[t]];
    sscan[t] = m;
    __syncthreads();
    for (int off = 1; off < NN; off <<= 1) {
        int v = (t >= off) ? sscan[t - off] : 0;
        __syncthreads();
        sscan[t] += v;
        __syncthreads();
    }
    const int nrem = (int)((float)n_nodes[b] * 0.5f);   // (1 - POOL_RATIO)
    const int rank = sscan[t] - m;
    const int nm   = (m == 1 && rank < nrem) ? 0 : m;
    new_mask_i[b * NN + sidx[t]] = nm;                  // scatter == inverse perm
    out_mask_f[b * NN + sidx[t]] = (float)nm;
    if (t == 0) out_nn_f[b] = (float)(n_nodes[b] - nrem);
}

// ------------------------------------------------------ elementwise outputs
__global__ void k_xout(const float* __restrict__ h,
                       const float* __restrict__ y,
                       const int* __restrict__ nm,
                       float* __restrict__ xout) {
    size_t i = (size_t)blockIdx.x * 256 + threadIdx.x;  // < 4194304
    size_t node = i >> 8;
    xout[i] = h[i] * tanhf(y[node]) * (float)nm[node];
}

__global__ void k_adjout(const float* __restrict__ adj,
                         const int* __restrict__ nm,
                         float* __restrict__ adjout) {
    size_t i = (size_t)blockIdx.x * 256 + threadIdx.x;  // < 16777216
    int    mcol = (int)(i & 1023);
    size_t bn   = i >> 10;                              // b*N + n
    int    b    = (int)(bn >> 10);
    float  mf_n = (float)nm[bn];
    float  mf_m = (float)nm[(size_t)b * NN + mcol];
    adjout[i] = mf_n * adj[i] * mf_m;
}

// ---------------------------------------------------------------- launcher
extern "C" void kernel_launch(void* const* d_in, const int* in_sizes, int n_in,
                              void* d_out, int out_size, void* d_ws, size_t ws_size,
                              hipStream_t stream) {
    const float* x       = (const float*)d_in[0];   // [16,1024,256]
    const float* adj     = (const float*)d_in[1];   // [16,1024,1024]
    const int*   mask    = (const int*)  d_in[2];   // [16,1024]
    const int*   n_nodes = (const int*)  d_in[3];   // [16]
    const float* W       = (const float*)d_in[4];   // [256,256]
    const float* bias    = (const float*)d_in[5];   // [256]
    const float* pooling = (const float*)d_in[6];   // [256,1]

    // workspace carve-up
    char* ws = (char*)d_ws;
    __bf16* xb   = (__bf16*)ws;  ws += (size_t)BB * NN * CIN * 2;   //  8 MB
    __bf16* adjb = (__bf16*)ws;  ws += (size_t)BB * NN * NN  * 2;   // 32 MB
    __bf16* wt   = (__bf16*)ws;  ws += (size_t)CIN * COUT * 2;      // 128 KB
    __bf16* h0t  = (__bf16*)ws;  ws += (size_t)BB * COUT * NN * 2;  //  8 MB
    float*  h    = (float*)ws;   ws += (size_t)BB * NN * COUT * 4;  // 16 MB
    float*  y    = (float*)ws;   ws += (size_t)BB * NN * 4;
    int*    nm   = (int*)ws;     ws += (size_t)BB * NN * 4;

    // output carve-up (flat f32, tuple return order)
    float* out     = (float*)d_out;
    float* xout    = out;                                       // 4194304
    float* adjout  = out + (size_t)BB * NN * COUT;              // 16777216
    float* maskf   = adjout + (size_t)BB * NN * NN;             // 16384
    float* nnf     = maskf + (size_t)BB * NN;                   // 16

    const int nx   = BB * NN * CIN;   // 4194304
    const int nadj = BB * NN * NN;    // 16777216

    k_f32_to_bf16<<<nx   / 256, 256, 0, stream>>>(x,   xb,   nx);
    k_f32_to_bf16<<<nadj / 256, 256, 0, stream>>>(adj, adjb, nadj);
    k_transpose_w<<<(CIN * COUT) / 256, 256, 0, stream>>>(W, wt);

    k_gemm1<<<2048, 256, 0, stream>>>(xb, wt, h0t);
    // 1024 blocks, each a 64x64 output tile with 266240B LDS (A+B K-bands)
    k_gemm2<<<1024, 256, 2 * LDS_TILE_BYTES, stream>>>(adjb, h0t, bias, h);

    k_score<<<BB * NN, 256, 0, stream>>>(h, pooling, y);
    k_pool<<<BB, NN, 0, stream>>>(y, mask, n_nodes, nm, maskf, nnf);

    k_xout<<<nx / 256, 256, 0, stream>>>(h, y, nm, xout);
    k_adjout<<<nadj / 256, 256, 0, stream>>>(adj, nm, adjout);
}
